// MultiScalePropagation_58067957842351
// MI455X (gfx1250) — matrix operand order repaired and checked by reading, not compiled, not run on hardware
//
#include <hip/hip_runtime.h>
#include <hip/hip_bf16.h>

// ---------------- problem constants ----------------
#define BB     4
#define NN_    16384
#define NPT    24
#define NSAMP  24
#define NSC    5
#define L1C    624      // 128+256+128+64+48
#define F1     512
#define F2     384
#define OCH    512
#define FPC    1139     // 512 (g) + 3 (xyz) + 624 (interp)
#define TP     64       // points per block in main kernel

typedef __attribute__((ext_vector_type(16))) _Float16 v16h;
typedef __attribute__((ext_vector_type(8)))  float    v8f;
typedef __attribute__((ext_vector_type(4)))  unsigned int u32x4;
typedef __attribute__((ext_vector_type(8)))  int      i32x8;
typedef __attribute__((ext_vector_type(4)))  int      i32x4;

__device__ __constant__ float d_rad2[NSC]  = {0.01f, 0.04f, 0.09f, 0.36f, 1.0f};
__device__ __constant__ int   d_mid[NSC]   = {64, 128, 64, 32, 24};
__device__ __constant__ int   d_oc[NSC]    = {128, 256, 128, 64, 48};
__device__ __constant__ int   d_choff[NSC] = {0, 128, 384, 512, 576};

struct SAParams {
    const float* w0[NSC]; const float* b0[NSC];
    const float* w1[NSC]; const float* b1[NSC];
};

// ---------------- 1) farthest point sampling ----------------
__global__ __launch_bounds__(1024)
void fps_kernel(const float* __restrict__ xyz, int* __restrict__ fps_idx,
                float* __restrict__ new_xyz) {
    const int b   = blockIdx.x;
    const int tid = threadIdx.x;
    const float* px = xyz + (size_t)b * 3 * NN_;
    __shared__ float sval[1024];
    __shared__ int   sidx[1024];
    float dist[NN_ / 1024];
    int far = 0;
    for (int it = 0; it < NPT; ++it) {
        if (tid == 0) {
            fps_idx[b * NPT + it] = far;
            new_xyz[(b * NPT + it) * 3 + 0] = px[0 * NN_ + far];
            new_xyz[(b * NPT + it) * 3 + 1] = px[1 * NN_ + far];
            new_xyz[(b * NPT + it) * 3 + 2] = px[2 * NN_ + far];
        }
        const float cx = px[0 * NN_ + far];
        const float cy = px[1 * NN_ + far];
        const float cz = px[2 * NN_ + far];
        float best = -1.0f; int bi = 0;
        #pragma unroll
        for (int i = 0; i < NN_ / 1024; ++i) {
            const int n = tid + i * 1024;
            const float dx = px[n] - cx, dy = px[NN_ + n] - cy, dz = px[2 * NN_ + n] - cz;
            const float d = dx * dx + dy * dy + dz * dz;
            dist[i] = (it == 0) ? d : fminf(dist[i], d);
            if (dist[i] > best) { best = dist[i]; bi = n; }
        }
        sval[tid] = best; sidx[tid] = bi;
        __syncthreads();
        for (int s = 512; s > 0; s >>= 1) {
            if (tid < s) {
                const float v = sval[tid + s]; const int j = sidx[tid + s];
                if (v > sval[tid] || (v == sval[tid] && j < sidx[tid])) {
                    sval[tid] = v; sidx[tid] = j;
                }
            }
            __syncthreads();
        }
        far = sidx[0];
        __syncthreads();
    }
}

// ---------------- 2) ball query ----------------
__global__ __launch_bounds__(128)
void ballq_kernel(const float* __restrict__ xyz, const float* __restrict__ new_xyz,
                  int* __restrict__ ball_idx) {
    const int blk = blockIdx.x;
    const int b   = blk / (NSC * NPT);
    const int rem = blk % (NSC * NPT);
    const int s   = rem / NPT;
    const int c   = rem % NPT;
    const int tid = threadIdx.x;
    const float r2 = d_rad2[s];
    const float* px = xyz + (size_t)b * 3 * NN_;
    const float cx = new_xyz[(b * NPT + c) * 3 + 0];
    const float cy = new_xyz[(b * NPT + c) * 3 + 1];
    const float cz = new_xyz[(b * NPT + c) * 3 + 2];
    __shared__ int pred[128];
    __shared__ int list[NSAMP];
    __shared__ int cnt;
    if (tid == 0) cnt = 0;
    __syncthreads();
    for (int base = 0; base < NN_; base += 128) {
        const int n = base + tid;
        const float dx = px[n] - cx, dy = px[NN_ + n] - cy, dz = px[2 * NN_ + n] - cz;
        pred[tid] = (dx * dx + dy * dy + dz * dz <= r2) ? 1 : 0;
        __syncthreads();
        if (tid == 0) {
            for (int i = 0; i < 128 && cnt < NSAMP; ++i)
                if (pred[i]) list[cnt++] = base + i;
        }
        __syncthreads();
        if (cnt >= NSAMP) break;
    }
    if (tid < NSAMP) {
        const int v = (tid < cnt) ? list[tid] : list[0];
        ball_idx[((b * NSC + s) * NPT + c) * NSAMP + tid] = v;
    }
}

// ---------------- 3) SA MLPs + maxpool -> l1_pts[b][center][624] ----------------
__global__ __launch_bounds__(256)
void sa_kernel(const float* __restrict__ xyz, const float* __restrict__ new_xyz,
               const int* __restrict__ ball_idx, float* __restrict__ l1_pts,
               SAParams P) {
    const int blk = blockIdx.x;
    const int b   = blk / (NSC * NPT);
    const int rem = blk % (NSC * NPT);
    const int s   = rem / NPT;
    const int c   = rem % NPT;
    const int tid = threadIdx.x;
    const int mid = d_mid[s], oc = d_oc[s];
    const float* px = xyz + (size_t)b * 3 * NN_;
    __shared__ float h0[NSAMP * 6];
    __shared__ float ha[NSAMP * 256];
    __shared__ float hb[NSAMP * 256];
    if (tid < NSAMP) {
        const int gi = ball_idx[((b * NSC + s) * NPT + c) * NSAMP + tid];
        const float ax = px[gi], ay = px[NN_ + gi], az = px[2 * NN_ + gi];
        const float cx = new_xyz[(b * NPT + c) * 3 + 0];
        const float cy = new_xyz[(b * NPT + c) * 3 + 1];
        const float cz = new_xyz[(b * NPT + c) * 3 + 2];
        h0[tid * 6 + 0] = ax;      h0[tid * 6 + 1] = ay;      h0[tid * 6 + 2] = az;
        h0[tid * 6 + 3] = ax - cx; h0[tid * 6 + 4] = ay - cy; h0[tid * 6 + 5] = az - cz;
    }
    __syncthreads();
    const float* W0 = P.w0[s]; const float* B0 = P.b0[s];
    for (int e = tid; e < NSAMP * mid; e += 256) {
        const int k = e / mid, o = e % mid;
        float sum = B0[o];
        #pragma unroll
        for (int cc = 0; cc < 6; ++cc) sum += W0[o * 6 + cc] * h0[k * 6 + cc];
        ha[k * mid + o] = fmaxf(sum, 0.0f);
    }
    __syncthreads();
    const float* W1 = P.w1[s]; const float* B1 = P.b1[s];
    for (int e = tid; e < NSAMP * oc; e += 256) {
        const int k = e / oc, o = e % oc;
        float sum = B1[o];
        for (int cc = 0; cc < mid; ++cc) sum += W1[o * mid + cc] * ha[k * mid + cc];
        hb[k * oc + o] = fmaxf(sum, 0.0f);
    }
    __syncthreads();
    for (int o = tid; o < oc; o += 256) {
        float m = -1e30f;
        for (int k = 0; k < NSAMP; ++k) m = fmaxf(m, hb[k * oc + o]);
        l1_pts[(size_t)(b * NPT + c) * L1C + d_choff[s] + o] = m;
    }
}

// ---------------- 4) per-batch folded constants ----------------
__global__ __launch_bounds__(512)
void c0_kernel(const float* __restrict__ g, const float* __restrict__ fpw1,
               const float* __restrict__ fpb1, float* __restrict__ c0) {
    const int b = blockIdx.x, o = threadIdx.x;
    const float* w = fpw1 + (size_t)o * FPC;
    const float* gp = g + (size_t)b * 512;
    float sum = fpb1[o];
    for (int cc = 0; cc < 512; ++cc) sum += w[cc] * gp[cc];
    c0[b * F1 + o] = sum;
}

__global__ void w1x_kernel(const float* __restrict__ fpw1, float* __restrict__ w1x) {
    const int t = blockIdx.x * blockDim.x + threadIdx.x;
    if (t < F1 * 3) {
        const int o = t / 3, j = t % 3;
        w1x[t] = fpw1[(size_t)o * FPC + 512 + j];
    }
}

__global__ __launch_bounds__(512)
void pc_kernel(const float* __restrict__ fpw1, const float* __restrict__ l1_pts,
               float* __restrict__ pc) {
    const int blk = blockIdx.x;
    const int b = blk / NPT, si = blk % NPT;
    const int o = threadIdx.x;
    const float* w = fpw1 + (size_t)o * FPC + 515;
    const float* l = l1_pts + (size_t)(b * NPT + si) * L1C;
    float sum = 0.0f;
    for (int cc = 0; cc < L1C; ++cc) sum += w[cc] * l[cc];
    pc[(size_t)(b * NPT + si) * F1 + o] = sum;
}

// ---------------- 5) 3-NN + normalized inverse-distance weights ----------------
__global__ __launch_bounds__(256)
void nn_kernel(const float* __restrict__ xyz, const float* __restrict__ new_xyz,
               int* __restrict__ nn_idx, float* __restrict__ nn_w) {
    const int b = blockIdx.y;
    const int n = blockIdx.x * 256 + threadIdx.x;
    __shared__ float cx[NPT], cy[NPT], cz[NPT];
    if (threadIdx.x < NPT) {
        cx[threadIdx.x] = new_xyz[(b * NPT + threadIdx.x) * 3 + 0];
        cy[threadIdx.x] = new_xyz[(b * NPT + threadIdx.x) * 3 + 1];
        cz[threadIdx.x] = new_xyz[(b * NPT + threadIdx.x) * 3 + 2];
    }
    __syncthreads();
    const float* px = xyz + (size_t)b * 3 * NN_;
    const float x = px[n], y = px[NN_ + n], z = px[2 * NN_ + n];
    float d0 = 1e30f, d1 = 1e30f, d2 = 1e30f;
    int i0 = 0, i1 = 0, i2 = 0;
    #pragma unroll
    for (int j = 0; j < NPT; ++j) {
        const float dx = x - cx[j], dy = y - cy[j], dz = z - cz[j];
        const float d = dx * dx + dy * dy + dz * dz;
        if (d < d0)      { d2 = d1; i2 = i1; d1 = d0; i1 = i0; d0 = d; i0 = j; }
        else if (d < d1) { d2 = d1; i2 = i1; d1 = d; i1 = j; }
        else if (d < d2) { d2 = d; i2 = j; }
    }
    float wa = 1.0f / (d0 + 1e-8f), wb = 1.0f / (d1 + 1e-8f), wc = 1.0f / (d2 + 1e-8f);
    const float inv = 1.0f / (wa + wb + wc);
    wa *= inv; wb *= inv; wc *= inv;
    const size_t base = (size_t)(b * NN_ + n) * 3;
    nn_idx[base + 0] = i0; nn_idx[base + 1] = i1; nn_idx[base + 2] = i2;
    nn_w[base + 0] = wa;  nn_w[base + 1] = wb;  nn_w[base + 2] = wc;
}

// ---------------- 6) pack weights to f16 WMMA B-fragment order ----------------
__global__ void pack_kernel(const float* __restrict__ W, _Float16* __restrict__ out,
                            int NT, int KS, int Kdim) {
    const int t = blockIdx.x * blockDim.x + threadIdx.x;
    const int total = NT * KS * 512;
    if (t >= total) return;
    const int h  = t & 15;
    const int L  = (t >> 4) & 31;
    const int f  = t >> 9;
    const int ks = f % KS;
    const int nt = f / KS;
    const int n  = nt * 16 + (L & 15);
    const int k  = ks * 32 + ((L >= 16) ? 8 : 0) + ((h < 8) ? h : (h + 8));
    out[t] = (_Float16)W[(size_t)n * Kdim + k];
}

// ---------------- 7) fused FP stage: interp-fold + two WMMA GEMMs ----------------
union AV { v16h v; float4 f[2]; };

__global__ __launch_bounds__(128)
void fp_main_kernel(const float* __restrict__ c0, const float* __restrict__ w1x,
                    const float* __restrict__ pc, const int* __restrict__ nn_idx,
                    const float* __restrict__ nn_w, const float* __restrict__ xyz,
                    const _Float16* __restrict__ pw2, const float* __restrict__ fpb2,
                    const _Float16* __restrict__ pwp, const float* __restrict__ projb,
                    float* __restrict__ out) {
    const int tid  = threadIdx.x;
    const int blk  = blockIdx.x;
    const int b    = blk / (NN_ / TP);
    const int tile = blk % (NN_ / TP);
    const int p0   = tile * TP;

    __shared__ __align__(16) _Float16 sh1[TP * F1];                  // 64 KB
    __shared__ __align__(16) unsigned char sreg[NPT * F1 * 4];       // 48 KB: pc (ph1) / h2 (ph2+)
    __shared__ float sc0[F1];
    __shared__ float sw1x[F1 * 3];
    __shared__ int   snn[TP * 3];
    __shared__ float snw[TP * 3];
    __shared__ float sxyz[TP * 3];
    float*    spc = (float*)sreg;      // [NPT*F1]
    _Float16* sh2 = (_Float16*)sreg;   // [TP*F2]  (NPT*F1*4 == TP*F2*2 == 49152)

    // ---- stage pc[b] (48 KB) into LDS with the Tensor Data Mover (wave 0 only;
    //      TDM ignores EXEC, so gate with a wave-uniform branch) ----
#if __has_builtin(__builtin_amdgcn_tensor_load_to_lds) && __has_builtin(__builtin_amdgcn_s_wait_tensorcnt)
    if (tid < 32) {
        const unsigned long long ga =
            (unsigned long long)(uintptr_t)(pc + (size_t)b * NPT * F1);
        const unsigned ldsa = (unsigned)(uintptr_t)spc;   // low 32 bits = LDS offset
        // D# group0: count=1 | lds_addr | global_addr[56:0] | type=2
        u32x4 g0 = { 1u, ldsa, (unsigned)ga,
                     (unsigned)((ga >> 32) & 0x01FFFFFFu) | (2u << 30) };
        // D# group1: data_size=4B (code 2); 1-D tile of NPT*F1 dwords
        i32x8 g1;
        g1[0] = (int)(2u << 16);                              // wg_mask=0, data_size=2
        g1[1] = (int)(((NPT * F1) & 0xFFFF) << 16);           // tensor_dim0 lo16 @ [63:48]
        g1[2] = (int)(((NPT * F1) >> 16) & 0xFFFF);           // tensor_dim0 hi16; tensor_dim1=0
        g1[3] = (int)(((NPT * F1) & 0xFFFF) << 16);           // tile_dim0 @ [127:112]
        g1[4] = 0;                                            // tile_dim1/2 unused
        g1[5] = (int)(NPT * F1);                              // tensor_dim0_stride lo32
        g1[6] = 0;
        g1[7] = 0;
        i32x4 gz4 = {0, 0, 0, 0};
        i32x8 gz8 = {0, 0, 0, 0, 0, 0, 0, 0};
        __builtin_amdgcn_tensor_load_to_lds(g0, g1, gz4, gz4, gz8, 0);
        __builtin_amdgcn_s_wait_tensorcnt(0);
    }
#else
    for (int i = tid; i < NPT * F1; i += 128) spc[i] = pc[(size_t)b * NPT * F1 + i];
#endif

    for (int i = tid; i < F1; i += 128)     sc0[i]  = c0[b * F1 + i];
    for (int i = tid; i < F1 * 3; i += 128) sw1x[i] = w1x[i];
    for (int i = tid; i < TP; i += 128) {
        const int p = p0 + i;
        const size_t base = (size_t)(b * NN_ + p) * 3;
        snn[i * 3 + 0] = nn_idx[base + 0];
        snn[i * 3 + 1] = nn_idx[base + 1];
        snn[i * 3 + 2] = nn_idx[base + 2];
        snw[i * 3 + 0] = nn_w[base + 0];
        snw[i * 3 + 1] = nn_w[base + 1];
        snw[i * 3 + 2] = nn_w[base + 2];
        const float* px = xyz + (size_t)b * 3 * NN_;
        sxyz[i * 3 + 0] = px[p];
        sxyz[i * 3 + 1] = px[NN_ + p];
        sxyz[i * 3 + 2] = px[2 * NN_ + p];
    }
    __syncthreads();

    // phase 1: h1 = relu(c0 + sum w_k * pc[nn_k] + W1xyz . xyz)
    for (int p = 0; p < TP; ++p) {
        const int   j0 = snn[p * 3], j1 = snn[p * 3 + 1], j2 = snn[p * 3 + 2];
        const float w0 = snw[p * 3], w1 = snw[p * 3 + 1], w2 = snw[p * 3 + 2];
        const float x = sxyz[p * 3], y = sxyz[p * 3 + 1], z = sxyz[p * 3 + 2];
        for (int c = tid; c < F1; c += 128) {
            float v = sc0[c]
                    + w0 * spc[j0 * F1 + c] + w1 * spc[j1 * F1 + c] + w2 * spc[j2 * F1 + c]
                    + x * sw1x[c * 3 + 0] + y * sw1x[c * 3 + 1] + z * sw1x[c * 3 + 2];
            sh1[p * F1 + c] = (_Float16)fmaxf(v, 0.0f);
        }
    }
    __syncthreads();

    const int wv    = tid >> 5;
    const int lane  = tid & 31;
    const int m0    = wv * 16;
    const int lm    = lane & 15;
    const int khalf = (lane >> 4) ? 8 : 0;       // A-matrix 16-bit layout K offset
    const int mhi   = (lane >> 4) ? 8 : 0;       // C/D layout M offset

    // phase 2: layer2 GEMM (K=512, N=384): two n-tiles per pass -> two
    // independent WMMA accumulation chains (hides XDL RAW latency)
    for (int nt = 0; nt < F2 / 16; nt += 2) {
        if (nt + 2 < F2 / 16)
            __builtin_prefetch(&pw2[((size_t)(nt + 2) * (F1 / 32) * 32 + lane) * 16], 0, 1);
        v8f acc0 = {0.f, 0.f, 0.f, 0.f, 0.f, 0.f, 0.f, 0.f};
        v8f acc1 = {0.f, 0.f, 0.f, 0.f, 0.f, 0.f, 0.f, 0.f};
        #pragma unroll
        for (int ks = 0; ks < F1 / 32; ++ks) {
            AV a;
            const _Float16* ap = &sh1[(m0 + lm) * F1 + ks * 32 + khalf];
            a.f[0] = *(const float4*)(ap);
            a.f[1] = *(const float4*)(ap + 16);
            const v16h b0 = *(const v16h*)&pw2[((size_t)(nt * (F1 / 32) + ks) * 32 + lane) * 16];
            const v16h b1 = *(const v16h*)&pw2[((size_t)((nt + 1) * (F1 / 32) + ks) * 32 + lane) * 16];
            acc0 = __builtin_amdgcn_wmma_f32_16x16x32_f16(false, a.v, false, b0,
                                                          (short)0, acc0, false, false);
            acc1 = __builtin_amdgcn_wmma_f32_16x16x32_f16(false, a.v, false, b1,
                                                          (short)0, acc1, false, false);
        }
        const int n0 = nt * 16 + lm;
        const int n1 = n0 + 16;
        const float bias0 = fpb2[n0];
        const float bias1 = fpb2[n1];
        #pragma unroll
        for (int r = 0; r < 8; ++r) {
            const int m = m0 + mhi + r;
            sh2[m * F2 + n0] = (_Float16)fmaxf(acc0[r] + bias0, 0.0f);
            sh2[m * F2 + n1] = (_Float16)fmaxf(acc1[r] + bias1, 0.0f);
        }
    }
    __syncthreads();

    // phase 3: proj GEMM (K=384, N=512) -> out (B, 512, N); dual chains again
    for (int nt = 0; nt < OCH / 16; nt += 2) {
        if (nt + 2 < OCH / 16)
            __builtin_prefetch(&pwp[((size_t)(nt + 2) * (F2 / 32) * 32 + lane) * 16], 0, 1);
        v8f acc0 = {0.f, 0.f, 0.f, 0.f, 0.f, 0.f, 0.f, 0.f};
        v8f acc1 = {0.f, 0.f, 0.f, 0.f, 0.f, 0.f, 0.f, 0.f};
        #pragma unroll
        for (int ks = 0; ks < F2 / 32; ++ks) {
            AV a;
            const _Float16* ap = &sh2[(m0 + lm) * F2 + ks * 32 + khalf];
            a.f[0] = *(const float4*)(ap);
            a.f[1] = *(const float4*)(ap + 16);
            const v16h b0 = *(const v16h*)&pwp[((size_t)(nt * (F2 / 32) + ks) * 32 + lane) * 16];
            const v16h b1 = *(const v16h*)&pwp[((size_t)((nt + 1) * (F2 / 32) + ks) * 32 + lane) * 16];
            acc0 = __builtin_amdgcn_wmma_f32_16x16x32_f16(false, a.v, false, b0,
                                                          (short)0, acc0, false, false);
            acc1 = __builtin_amdgcn_wmma_f32_16x16x32_f16(false, a.v, false, b1,
                                                          (short)0, acc1, false, false);
        }
        const int o0 = nt * 16 + lm;
        const int o1 = o0 + 16;
        const float bias0 = projb[o0];
        const float bias1 = projb[o1];
        #pragma unroll
        for (int r = 0; r < 8; ++r) {
            const int pnt = p0 + m0 + mhi + r;
            out[((size_t)b * OCH + o0) * NN_ + pnt] = acc0[r] + bias0;
            out[((size_t)b * OCH + o1) * NN_ + pnt] = acc1[r] + bias1;
        }
    }
}

// ---------------- launch ----------------
extern "C" void kernel_launch(void* const* d_in, const int* in_sizes, int n_in,
                              void* d_out, int out_size, void* d_ws, size_t ws_size,
                              hipStream_t stream) {
    const float* xyz  = (const float*)d_in[0];   // (4, 3, 16384)
    const float* g    = (const float*)d_in[1];   // (4, 512, 1)
    SAParams P;
    for (int s = 0; s < NSC; ++s) {
        P.w0[s] = (const float*)d_in[2 + s * 4 + 0];
        P.b0[s] = (const float*)d_in[2 + s * 4 + 1];
        P.w1[s] = (const float*)d_in[2 + s * 4 + 2];
        P.b1[s] = (const float*)d_in[2 + s * 4 + 3];
    }
    const float* fpw1  = (const float*)d_in[22]; // (512, 1139)
    const float* fpb1  = (const float*)d_in[23]; // (512)
    const float* fpw2  = (const float*)d_in[24]; // (384, 512)
    const float* fpb2  = (const float*)d_in[25]; // (384)
    const float* projw = (const float*)d_in[26]; // (512, 384)
    const float* projb = (const float*)d_in[27]; // (512)
    float* out = (float*)d_out;

    char* ws = (char*)d_ws;
    size_t off = 0;
    auto carve = [&](size_t bytes) -> char* {
        char* p = ws + off;
        off = (off + bytes + 255) & ~(size_t)255;
        return p;
    };
    int*      fps_idx  = (int*)   carve((size_t)BB * NPT * sizeof(int));
    float*    new_xyz  = (float*) carve((size_t)BB * NPT * 3 * sizeof(float));
    int*      ball_idx = (int*)   carve((size_t)BB * NSC * NPT * NSAMP * sizeof(int));
    float*    l1_pts   = (float*) carve((size_t)BB * NPT * L1C * sizeof(float));
    float*    c0       = (float*) carve((size_t)BB * F1 * sizeof(float));
    float*    w1x      = (float*) carve((size_t)F1 * 3 * sizeof(float));
    float*    pc       = (float*) carve((size_t)BB * NPT * F1 * sizeof(float));
    int*      nn_idx   = (int*)   carve((size_t)BB * NN_ * 3 * sizeof(int));
    float*    nn_w     = (float*) carve((size_t)BB * NN_ * 3 * sizeof(float));
    _Float16* pw2      = (_Float16*)carve((size_t)F2 * F1 * sizeof(_Float16));
    _Float16* pwp      = (_Float16*)carve((size_t)OCH * F2 * sizeof(_Float16));
    (void)ws_size; (void)in_sizes; (void)n_in; (void)out_size;

    fps_kernel  <<<BB, 1024, 0, stream>>>(xyz, fps_idx, new_xyz);
    ballq_kernel<<<BB * NSC * NPT, 128, 0, stream>>>(xyz, new_xyz, ball_idx);
    sa_kernel   <<<BB * NSC * NPT, 256, 0, stream>>>(xyz, new_xyz, ball_idx, l1_pts, P);
    c0_kernel   <<<BB, 512, 0, stream>>>(g, fpw1, fpb1, c0);
    w1x_kernel  <<<6, 256, 0, stream>>>(fpw1, w1x);
    pc_kernel   <<<BB * NPT, 512, 0, stream>>>(fpw1, l1_pts, pc);
    nn_kernel   <<<dim3(NN_ / 256, BB), 256, 0, stream>>>(xyz, new_xyz, nn_idx, nn_w);
    pack_kernel <<<(F2 / 16) * (F1 / 32) * 2, 256, 0, stream>>>(fpw2, pw2, F2 / 16, F1 / 32, F1);
    pack_kernel <<<(OCH / 16) * (F2 / 32) * 2, 256, 0, stream>>>(projw, pwp, OCH / 16, F2 / 32, F2);
    fp_main_kernel<<<BB * (NN_ / TP), 128, 0, stream>>>(c0, w1x, pc, nn_idx, nn_w, xyz,
                                                        pw2, fpb2, pwp, projb, out);
}
